// SpecformerSmall_21887153340957
// MI455X (gfx1250) — matrix operand order repaired and checked by reading, not compiled, not run on hardware
//
#include <hip/hip_runtime.h>
#include <math.h>

// ---------------- model constants ----------------
#define BB   8
#define NN   512
#define HH   256
#define NHD  8       // heads
#define HDIM 32      // head dim
#define NL   4
#define NC   10
#define DEG  32
#define NE   (BB * NN * DEG)      // 131072 edges
#define NNODE (BB * NN)           // 4096 nodes

typedef __attribute__((ext_vector_type(16))) __bf16 v16bf;
typedef __attribute__((ext_vector_type(8)))  float  v8f;
typedef int v4i_vec __attribute__((vector_size(16)));

#if __has_builtin(__builtin_amdgcn_global_load_async_to_lds_b128)
#define HAS_ASYNC_LDS 1
#else
#define HAS_ASYNC_LDS 0
#endif

// ---------------- device helpers ----------------
__device__ __forceinline__ float gelu_exact(float x) {
    return 0.5f * x * (1.f + erff(x * 0.70710678118654752f));
}

__device__ __forceinline__ void wait_async_then_barrier() {
#if HAS_ASYNC_LDS
#if __has_builtin(__builtin_amdgcn_s_wait_asynccnt)
    __builtin_amdgcn_s_wait_asynccnt(0);
#else
    asm volatile("s_wait_asynccnt 0x0" ::: "memory");
#endif
#endif
    __syncthreads();
}

// =====================================================================
// FAST-PATH WMMA GEMM (D == 256, M % 128 == 0, O % 64 == 0):
//   Out[M,O] = act(BN(A[M,256] @ W[O,256]^T + bias)) (+resid)
// Workgroup = 8 waves -> 128x64 output tile. The 64x256 f32 weight panel
// is staged into LDS with global_load_async_to_lds_b128 (CDNA5 async path),
// then each wave computes 16 rows x 4 col-tiles, reusing its A fragment
// across 4 v_wmma per k-step. Fully unrolled K: 32 wmma per wave.
// =====================================================================
template <bool ABF16, bool OBF16>
__global__ void wmma_gemm_staged(const void* __restrict__ Ap,
                                 const float* __restrict__ W,
                                 const float* __restrict__ bias,
                                 const float* __restrict__ bng,
                                 const float* __restrict__ bnb,
                                 const float* __restrict__ bnm,
                                 const float* __restrict__ bnv,
                                 const float* __restrict__ resid,
                                 void* __restrict__ Outp,
                                 int M, int O, int act) {
    __shared__ __align__(16) float Wlds[64 * HH];   // 64 KB panel

    const int obct = O >> 6;
    const int mb   = blockIdx.x / obct;
    const int ob   = blockIdx.x - mb * obct;
    const int t    = threadIdx.x;

    // ---- stage W panel [64 x 256] into LDS ----
    const float* wpanel = W + (size_t)(ob * 64) * HH;
    #pragma unroll
    for (int i = 0; i < 16; ++i) {
        const int chunk = i * 256 + t;              // 16-byte chunks
#if HAS_ASYNC_LDS
        __builtin_amdgcn_global_load_async_to_lds_b128(
            (__attribute__((address_space(1))) v4i_vec*)(wpanel + chunk * 4),
            (__attribute__((address_space(3))) v4i_vec*)(Wlds + chunk * 4),
            0, 0);
#else
        *(float4*)(Wlds + chunk * 4) = *(const float4*)(wpanel + chunk * 4);
#endif
    }
    wait_async_then_barrier();

    const int wv   = t >> 5;
    const int lane = t & 31;
    const int r    = lane & 15;
    const int half = lane >> 4;

    const int arow = mb * 128 + wv * 16 + r;
    const float*  __restrict__ Af = (const float*)Ap;
    const __bf16* __restrict__ Ab = (const __bf16*)Ap;

    v8f acc[4] = {};
    #pragma unroll
    for (int kk = 0; kk < HH; kk += 32) {
        // A fragment (contiguous k order): lane<16 K{kk..+15}, lane>=16 K{kk+16..+31}
        v16bf a;
        const int ka = kk + half * 16;
        #pragma unroll
        for (int i = 0; i < 8; ++i) {
            float a0, a1;
            if (ABF16) {
                a0 = (float)Ab[(size_t)arow * HH + ka + i];
                a1 = (float)Ab[(size_t)arow * HH + ka + 8 + i];
            } else {
                a0 = Af[(size_t)arow * HH + ka + i];
                a1 = Af[(size_t)arow * HH + ka + 8 + i];
            }
            a[i]     = (__bf16)a0;
            a[i + 8] = (__bf16)a1;
        }
        // 4 B fragments from the LDS panel, 4 wmma reusing `a`
        #pragma unroll
        for (int tc = 0; tc < 4; ++tc) {
            v16bf w;
            const float* wl = Wlds + (tc * 16 + r) * HH + kk + half * 16;
            #pragma unroll
            for (int i = 0; i < 16; ++i) w[i] = (__bf16)wl[i];
            acc[tc] = __builtin_amdgcn_wmma_f32_16x16x32_bf16(
                          false, a, false, w, (short)0, acc[tc], false, false);
        }
    }

    // ---- epilogue: col = ob*64 + tc*16 + r, rows = base + half*8 + j ----
    #pragma unroll
    for (int tc = 0; tc < 4; ++tc) {
        const int col = ob * 64 + tc * 16 + r;
        float gs = 1.f, gsh = 0.f;
        if (bng) {
            const float inv = rsqrtf(bnv[col] + 1e-5f);
            gs  = bng[col] * inv;
            gsh = bnb[col] - bnm[col] * gs;
        }
        const float bv = bias ? bias[col] : 0.f;
        #pragma unroll
        for (int j = 0; j < 8; ++j) {
            const int row = mb * 128 + wv * 16 + half * 8 + j;
            float v = acc[tc][j] + bv;
            if (bng) v = v * gs + gsh;
            if (act == 1) v = gelu_exact(v);
            else if (act == 2) v = fmaxf(v, 0.f);
            if (resid) v += resid[(size_t)row * O + col];
            if (OBF16) ((__bf16*)Outp)[(size_t)row * O + col] = (__bf16)v;
            else       ((float*)Outp)[(size_t)row * O + col] = v;
        }
    }
}

// =====================================================================
// Generic guarded WMMA GEMM for odd shapes (D=257/9, O=8/10, M=8).
// =====================================================================
template <bool OBF16>
__global__ void wmma_gemm_generic(const float* __restrict__ Af,
                                  const float* __restrict__ W,
                                  const float* __restrict__ bias,
                                  const float* __restrict__ bng,
                                  const float* __restrict__ bnb,
                                  const float* __restrict__ bnm,
                                  const float* __restrict__ bnv,
                                  const float* __restrict__ resid,
                                  void* __restrict__ Outp,
                                  int M, int D, int O, int act) {
    const int wave = (blockIdx.x * blockDim.x + threadIdx.x) >> 5;
    const int lane = threadIdx.x & 31;
    const int r    = lane & 15;
    const int half = lane >> 4;
    const int OT = (O + 15) >> 4;
    const int MT = (M + 15) >> 4;
    const int tm = wave / OT;
    const int to = wave - tm * OT;
    if (tm >= MT) return;

    const int arow = tm * 16 + r;
    const int wrow = to * 16 + r;
    const bool rv = arow < M;
    const bool cv = wrow < O;

    v8f acc = {};
    for (int kk = 0; kk < D; kk += 32) {
        if (kk + 32 < D)
            __builtin_prefetch(&W[(size_t)(cv ? wrow : 0) * D + kk + 32], 0, 0);
        v16bf a, w;
        const int ka = kk + half * 16;
        #pragma unroll
        for (int i = 0; i < 8; ++i) {
            const int k0 = ka + i, k1 = ka + 8 + i;
            const float a0 = (rv && k0 < D) ? Af[(size_t)arow * D + k0] : 0.f;
            const float a1 = (rv && k1 < D) ? Af[(size_t)arow * D + k1] : 0.f;
            a[i]     = (__bf16)a0;
            a[i + 8] = (__bf16)a1;
        }
        const int kb = kk + half * 16;
        #pragma unroll
        for (int i = 0; i < 16; ++i) {
            const int k = kb + i;
            w[i] = (__bf16)((cv && k < D) ? W[(size_t)wrow * D + k] : 0.f);
        }
        acc = __builtin_amdgcn_wmma_f32_16x16x32_bf16(
                  false, a, false, w, (short)0, acc, false, false);
    }

    const int col = to * 16 + r;
    if (col >= O) return;
    float gs = 1.f, gsh = 0.f;
    if (bng) {
        const float inv = rsqrtf(bnv[col] + 1e-5f);
        gs  = bng[col] * inv;
        gsh = bnb[col] - bnm[col] * gs;
    }
    const float bv = bias ? bias[col] : 0.f;
    #pragma unroll
    for (int j = 0; j < 8; ++j) {
        const int row = tm * 16 + half * 8 + j;
        if (row >= M) continue;
        float v = acc[j] + bv;
        if (bng) v = v * gs + gsh;
        if (act == 1) v = gelu_exact(v);
        else if (act == 2) v = fmaxf(v, 0.f);
        if (resid) v += resid[(size_t)row * O + col];
        if (OBF16) ((__bf16*)Outp)[(size_t)row * O + col] = (__bf16)v;
        else       ((float*)Outp)[(size_t)row * O + col] = v;
    }
}

// =====================================================================
// Sine encoding: eeig[row, 0..256] = [e, sin(100*e*div), cos(100*e*div)]
// =====================================================================
__global__ void sine_encode_kernel(const float* __restrict__ e, float* __restrict__ eeig) {
    const int tid = blockIdx.x * blockDim.x + threadIdx.x;   // NNODE*128
    const int row = tid >> 7;
    const int i   = tid & 127;
    const float ev = e[row];
    const float dv = __expf((float)(2 * i) * (-9.210340371976184f / 256.f));
    const float pe = ev * 100.f * dv;
    float* rr = eeig + (size_t)row * 257;
    if (i == 0) rr[0] = ev;
    rr[1 + i]   = __sinf(pe);
    rr[129 + i] = __cosf(pe);
}

// =====================================================================
// LayerNorm over last dim (256), one wave per row
// =====================================================================
__global__ void layernorm_kernel(const float* __restrict__ x,
                                 const float* __restrict__ g,
                                 const float* __restrict__ b,
                                 float* __restrict__ out, int M) {
    const int row  = (blockIdx.x * blockDim.x + threadIdx.x) >> 5;
    const int lane = threadIdx.x & 31;
    if (row >= M) return;
    const float* xr = x + (size_t)row * HH;
    float v[8];
    float s = 0.f;
    #pragma unroll
    for (int i = 0; i < 8; ++i) { v[i] = xr[lane + 32 * i]; s += v[i]; }
    #pragma unroll
    for (int off = 16; off; off >>= 1) s += __shfl_xor(s, off, 32);
    const float mu = s * (1.f / HH);
    float q = 0.f;
    #pragma unroll
    for (int i = 0; i < 8; ++i) { const float d = v[i] - mu; q += d * d; }
    #pragma unroll
    for (int off = 16; off; off >>= 1) q += __shfl_xor(q, off, 32);
    const float inv = rsqrtf(q * (1.f / HH) + 1e-5f);
    float* orow = out + (size_t)row * HH;
    #pragma unroll
    for (int i = 0; i < 8; ++i) {
        const int c = lane + 32 * i;
        orow[c] = (v[i] - mu) * inv * g[c] + b[c];
    }
}

// =====================================================================
// MHA with WMMA. One wave per (b, h, 16-query tile). hd=32 == WMMA K, so
// each 16x16 score tile is exactly one v_wmma_f32_16x16x32_bf16.
// qkv row layout: [q(256) | k(256) | v(256)], head h at offset h*32.
// =====================================================================
__global__ void attention_kernel(const float* __restrict__ qkv,
                                 float* __restrict__ attno) {
    const int blk  = blockIdx.x;
    const int qt   = blk & 31;
    const int h    = (blk >> 5) & 7;
    const int b    = blk >> 8;
    const int lane = threadIdx.x;
    const int r    = lane & 15;
    const int half = lane >> 4;

    __shared__ float  S[16][NN];
    __shared__ __bf16 P[16][NN];

    v16bf qa;
    const float* qb = qkv + ((size_t)(b * NN + qt * 16 + r)) * 768 + h * HDIM;
    #pragma unroll
    for (int i = 0; i < 8; ++i) {
        qa[i]     = (__bf16)qb[half * 16 + i];
        qa[i + 8] = (__bf16)qb[half * 16 + 8 + i];
    }
    const float scale = 0.17677669529663687f;   // 1/sqrt(32)

    for (int kt = 0; kt < NN / 16; ++kt) {
        v16bf kb;
        const float* kp = qkv + ((size_t)(b * NN + kt * 16 + r)) * 768
                              + HH + h * HDIM + half * 16;
        #pragma unroll
        for (int i = 0; i < 16; ++i) kb[i] = (__bf16)kp[i];
        v8f s = {};
        s = __builtin_amdgcn_wmma_f32_16x16x32_bf16(
                false, qa, false, kb, (short)0, s, false, false);
        #pragma unroll
        for (int j = 0; j < 8; ++j) S[half * 8 + j][kt * 16 + r] = s[j] * scale;
    }
    __syncthreads();

    if (lane < 16) {
        float mx = -1e30f;
        for (int k = 0; k < NN; ++k) mx = fmaxf(mx, S[lane][k]);
        float sum = 0.f;
        for (int k = 0; k < NN; ++k) { const float p = __expf(S[lane][k] - mx); sum += p; S[lane][k] = p; }
        const float inv = 1.f / sum;
        for (int k = 0; k < NN; ++k) P[lane][k] = (__bf16)(S[lane][k] * inv);
    }
    __syncthreads();

    for (int ot = 0; ot < 2; ++ot) {
        v8f acc = {};
        for (int kk = 0; kk < NN; kk += 32) {
            v16bf pa;
            #pragma unroll
            for (int i = 0; i < 8; ++i) {
                pa[i]     = P[r][kk + half * 16 + i];
                pa[i + 8] = P[r][kk + half * 16 + 8 + i];
            }
            v16bf vb;
            const float* vp = qkv + ((size_t)(b * NN + kk + half * 16)) * 768
                                  + 2 * HH + h * HDIM + ot * 16 + r;
            #pragma unroll
            for (int i = 0; i < 16; ++i) vb[i] = (__bf16)vp[(size_t)i * 768];
            acc = __builtin_amdgcn_wmma_f32_16x16x32_bf16(
                      false, pa, false, vb, (short)0, acc, false, false);
        }
        #pragma unroll
        for (int j = 0; j < 8; ++j)
            attno[((size_t)(b * NN + qt * 16 + half * 8 + j)) * HH
                  + h * HDIM + ot * 16 + r] = acc[j];
    }
}

// =====================================================================
// Edge spectral filters: be[e] = [src==dst, sum_k u[b,s,k]*u[b,d,k]*new_e[b,k,h]]
// =====================================================================
__global__ void edge_bases_kernel(const float* __restrict__ u,
                                  const float* __restrict__ new_e,
                                  const int* __restrict__ esrc,
                                  const int* __restrict__ edst,
                                  const int* __restrict__ ebatch,
                                  float* __restrict__ be) {
    const int e    = (blockIdx.x * blockDim.x + threadIdx.x) >> 5;
    const int lane = threadIdx.x & 31;
    const int b = ebatch[e], s = esrc[e], d = edst[e];
    const float* us = u + ((size_t)b * NN + s) * NN;
    const float* ud = u + ((size_t)b * NN + d) * NN;
    float acc[NHD] = {};
    for (int k = lane; k < NN; k += 32) {
        const float p = us[k] * ud[k];
        const float* ne = new_e + ((size_t)(b * NN + k)) * NHD;
        #pragma unroll
        for (int h = 0; h < NHD; ++h) acc[h] += p * ne[h];
    }
    #pragma unroll
    for (int h = 0; h < NHD; ++h)
        #pragma unroll
        for (int off = 16; off; off >>= 1) acc[h] += __shfl_xor(acc[h], off, 32);
    if (lane == 0) {
        be[(size_t)e * 9] = (s == d) ? 1.f : 0.f;
        #pragma unroll
        for (int h = 0; h < NHD; ++h) be[(size_t)e * 9 + 1 + h] = acc[h];
    }
}

// =====================================================================
// Message + scatter: x[gdst] += pre[gsrc] * fe   (native f32 global atomics)
// =====================================================================
__global__ void msg_scatter_kernel(const float* __restrict__ pre,
                                   const __bf16* __restrict__ fe,
                                   const int* __restrict__ esrc,
                                   const int* __restrict__ edst,
                                   const int* __restrict__ ebatch,
                                   float* __restrict__ x) {
    const size_t tid = (size_t)blockIdx.x * blockDim.x + threadIdx.x;
    const size_t e = tid >> 8;
    const int f = (int)(tid & 255);
    const int b = ebatch[e];
    const int gs = b * NN + esrc[e];
    const int gd = b * NN + edst[e];
    const float m = pre[(size_t)gs * HH + f] * (float)fe[e * HH + f];
    atomicAdd(&x[(size_t)gd * HH + f], m);
}

// =====================================================================
// Mean pool over the N dim
// =====================================================================
__global__ void pool_kernel(const float* __restrict__ x, float* __restrict__ pooled) {
    const int tid = blockIdx.x * blockDim.x + threadIdx.x;  // BB*HH
    const int b = tid >> 8, c = tid & 255;
    float s = 0.f;
    for (int n = 0; n < NN; ++n) s += x[((size_t)(b * NN + n)) * HH + c];
    pooled[b * HH + c] = s * (1.f / NN);
}

// ---------------- host side ----------------
static void gemm(hipStream_t st, bool abf, bool obf,
                 const void* A, const float* W, const float* bias,
                 const float* g, const float* bb, const float* m, const float* v,
                 const float* resid, void* out, int M, int D, int O, int act) {
    if (D == HH && (M & 127) == 0 && (O & 63) == 0) {
        const int blocks = (M >> 7) * (O >> 6);
        if (abf)
            wmma_gemm_staged<true, true><<<blocks, 256, 0, st>>>(
                A, W, bias, g, bb, m, v, resid, out, M, O, act);
        else
            wmma_gemm_staged<false, false><<<blocks, 256, 0, st>>>(
                A, W, bias, g, bb, m, v, resid, out, M, O, act);
    } else {
        const int MT = (M + 15) / 16, OT = (O + 15) / 16;
        const int blocks = (MT * OT + 7) / 8;
        if (obf)
            wmma_gemm_generic<true><<<blocks, 256, 0, st>>>(
                (const float*)A, W, bias, g, bb, m, v, resid, out, M, D, O, act);
        else
            wmma_gemm_generic<false><<<blocks, 256, 0, st>>>(
                (const float*)A, W, bias, g, bb, m, v, resid, out, M, D, O, act);
    }
}

enum {
    IN_E = 0, IN_U, IN_NODE, IN_SRC, IN_DST, IN_EBATCH,
    P_EIG_W, P_EIG_B,
    P_MHA_INW, P_MHA_INB, P_MHA_OUTW, P_MHA_OUTB,
    P_MHAN_G, P_MHAN_B, P_FFNN_G, P_FFNN_B,
    P_FFN1_W, P_FFN1_B, P_FFN2_W, P_FFN2_B,
    P_DEC_W, P_DEC_B,
    P_FE1_W, P_FE1_B, P_FBN1_G, P_FBN1_B, P_FBN1_M, P_FBN1_V,
    P_FE2_W, P_FE2_B, P_FBN2_G, P_FBN2_B, P_FBN2_M, P_FBN2_V,
    P_CONV0 = 34,             // 14 tensors per conv layer
    P_OUT_W = 34 + 4 * 14,    // 90
    P_OUT_B = 91
};

extern "C" void kernel_launch(void* const* d_in, const int* in_sizes, int n_in,
                              void* d_out, int out_size, void* d_ws, size_t ws_size,
                              hipStream_t stream) {
    (void)in_sizes; (void)n_in; (void)out_size; (void)ws_size;
    const float* e_in   = (const float*)d_in[IN_E];
    const float* u_in   = (const float*)d_in[IN_U];
    const float* nfeat  = (const float*)d_in[IN_NODE];
    const int*   esrc   = (const int*)d_in[IN_SRC];
    const int*   edst   = (const int*)d_in[IN_DST];
    const int*   ebatch = (const int*)d_in[IN_EBATCH];
    #define PF(i) ((const float*)d_in[(i)])

    // workspace carve-up
    char* base = (char*)d_ws;
    size_t off = 0;
    #define WS(ty, nelem) ([&]() -> ty* { off = (off + 255) & ~(size_t)255; \
        ty* p = (ty*)(base + off); off += (size_t)(nelem) * sizeof(ty); return p; }())
    float*  eeig   = WS(float,  (size_t)NNODE * 257);
    float*  eig    = WS(float,  (size_t)NNODE * HH);
    float*  xn     = WS(float,  (size_t)NNODE * HH);
    float*  qkv    = WS(float,  (size_t)NNODE * 768);
    float*  attno  = WS(float,  (size_t)NNODE * HH);
    float*  ffnh   = WS(float,  (size_t)NNODE * HH);
    float*  new_e  = WS(float,  (size_t)NNODE * NHD);
    float*  be     = WS(float,  (size_t)NE * 9);
    __bf16* h1     = WS(__bf16, (size_t)NE * HH);
    __bf16* fe     = WS(__bf16, (size_t)NE * HH);
    float*  prebuf = WS(float,  (size_t)NNODE * HH);
    float*  ybuf   = WS(float,  (size_t)NNODE * HH);
    float*  x      = WS(float,  (size_t)NNODE * HH);
    float*  pooled = WS(float,  (size_t)BB * HH);
    #undef WS

    // 1. sine encoding -> eeig [4096,257]
    sine_encode_kernel<<<(NNODE * 128) / 256, 256, 0, stream>>>(e_in, eeig);
    // 2. eig = eeig @ eig_w^T + b   [4096,257]->[4096,256]   (odd D -> generic)
    gemm(stream, false, false, eeig, PF(P_EIG_W), PF(P_EIG_B),
         nullptr, nullptr, nullptr, nullptr, nullptr, eig, NNODE, 257, HH, 0);
    // 3. pre-norm MHA
    layernorm_kernel<<<(NNODE * 32) / 256, 256, 0, stream>>>(eig, PF(P_MHAN_G), PF(P_MHAN_B), xn, NNODE);
    gemm(stream, false, false, xn, PF(P_MHA_INW), PF(P_MHA_INB),
         nullptr, nullptr, nullptr, nullptr, nullptr, qkv, NNODE, HH, 768, 0);
    attention_kernel<<<BB * NHD * (NN / 16), 32, 0, stream>>>(qkv, attno);
    gemm(stream, false, false, attno, PF(P_MHA_OUTW), PF(P_MHA_OUTB),
         nullptr, nullptr, nullptr, nullptr, eig, eig, NNODE, HH, HH, 0);
    // 4. FFN
    layernorm_kernel<<<(NNODE * 32) / 256, 256, 0, stream>>>(eig, PF(P_FFNN_G), PF(P_FFNN_B), xn, NNODE);
    gemm(stream, false, false, xn, PF(P_FFN1_W), PF(P_FFN1_B),
         nullptr, nullptr, nullptr, nullptr, nullptr, ffnh, NNODE, HH, HH, 1);
    gemm(stream, false, false, ffnh, PF(P_FFN2_W), PF(P_FFN2_B),
         nullptr, nullptr, nullptr, nullptr, eig, eig, NNODE, HH, HH, 0);
    // 5. decoder -> new_e [4096,8]   (O=8 -> generic)
    gemm(stream, false, false, eig, PF(P_DEC_W), PF(P_DEC_B),
         nullptr, nullptr, nullptr, nullptr, nullptr, new_e, NNODE, HH, NHD, 0);
    // 6. edge-gathered spectral filters -> be [E,9]
    edge_bases_kernel<<<NE / 8, 256, 0, stream>>>(u_in, new_e, esrc, edst, ebatch, be);
    // 7. fenc: GEMM+BN+GELU x2, outputs bf16 (L2-resident 67MB each)
    gemm(stream, false, true, be, PF(P_FE1_W), PF(P_FE1_B),
         PF(P_FBN1_G), PF(P_FBN1_B), PF(P_FBN1_M), PF(P_FBN1_V),
         nullptr, h1, NE, 9, HH, 1);
    gemm(stream, true, true, h1, PF(P_FE2_W), PF(P_FE2_B),
         PF(P_FBN2_G), PF(P_FBN2_B), PF(P_FBN2_M), PF(P_FBN2_V),
         nullptr, fe, NE, HH, HH, 1);
    // 8. GNN conv layers
    (void)hipMemcpyAsync(x, nfeat, (size_t)NNODE * HH * sizeof(float),
                         hipMemcpyDeviceToDevice, stream);
    for (int l = 0; l < NL; ++l) {
        const int c = P_CONV0 + l * 14;
        gemm(stream, false, false, x, PF(c + 0), PF(c + 1),
             nullptr, nullptr, nullptr, nullptr, nullptr, prebuf, NNODE, HH, HH, 1);
        msg_scatter_kernel<<<NE, 256, 0, stream>>>(prebuf, fe, esrc, edst, ebatch, x);
        gemm(stream, false, false, x, PF(c + 2), PF(c + 3),
             PF(c + 4), PF(c + 5), PF(c + 6), PF(c + 7),
             nullptr, ybuf, NNODE, HH, HH, 2);
        gemm(stream, false, false, ybuf, PF(c + 8), PF(c + 9),
             PF(c + 10), PF(c + 11), PF(c + 12), PF(c + 13),
             x, x, NNODE, HH, HH, 2);
    }
    // 9. pool + classifier -> d_out [8,10]
    pool_kernel<<<BB, 256, 0, stream>>>(x, pooled);
    gemm(stream, false, false, pooled, PF(P_OUT_W), PF(P_OUT_B),
         nullptr, nullptr, nullptr, nullptr, nullptr, d_out, BB, HH, NC, 0);
    #undef PF
}
